// PointTransformerClassif_32538672234531
// MI455X (gfx1250) — matrix operand order repaired and checked by reading, not compiled, not run on hardware
//
#include <hip/hip_runtime.h>

// ============================================================================
// PointTransformer classifier forward for gfx1250 (MI455X, wave32, WMMA).
// All linear layers run through a v_wmma_f32_16x16x32_f16 tiled GEMM
// (f16 inputs, f32 accumulate); glue ops are fused elementwise kernels.
// GEMM inner loops are branch-free (clamped addressing, no exec predication)
// and M-register-blocked: MT M-tiles per wave share one B fragment, so the
// 4 WMMAs issue back-to-back after one clause of loads. Addresses are carried
// as integer offsets from the kernarg base so loads lower to GLOBAL (not FLAT).
// ============================================================================

typedef __attribute__((ext_vector_type(16))) _Float16 v16h;
typedef __attribute__((ext_vector_type(8)))  float    v8f;

#define BB 4
#define NN 1024
#define DM 512
#define CDIVU(a,b) ((unsigned)(((size_t)(a) + (b) - 1) / (b)))

__device__ __forceinline__ int imin(int a, int b) { return a < b ? a : b; }

// ---------------------------------------------------------------------------
// Fast WMMA GEMM: requires K % 32 == 0 (no K-tail). Branch-free interior:
// out-of-range rows/cols are clamped (their results are never stored).
// Each wave computes MT vertically-adjacent 16x16 D tiles sharing one B
// fragment per K-step. A fragment = 4x float4 contiguous loads per tile;
// B fragment = 16 strided b32 loads. Layouts per CDNA5 ISA 7.12.2.
// ---------------------------------------------------------------------------
template <int ACT, int MT>
__global__ void wmma_gemm_fast_kernel(const float* __restrict__ X,
                                      const float* __restrict__ W,
                                      const float* __restrict__ bias,
                                      const float* __restrict__ resid,
                                      float* __restrict__ Y,
                                      int M, int K, int N)
{
    const int lane = threadIdx.x & 31;
    const int wave = threadIdx.x >> 5;
    const int tn   = blockIdx.x;
    const int tm0  = (blockIdx.y * 4 + wave) * MT;   // first M tile of this wave
    const int g    = lane >> 4;     // 0: lanes 0-15, 1: lanes 16-31
    const int r    = lane & 15;
    const int col  = tn * 16 + r;
    const int col_c = imin(col, N - 1);              // clamp: OOB lanes compute
                                                     // garbage that's never stored
    // Integer offsets (no pointer PHIs -> loads infer GLOBAL addrspace).
    size_t aoff[MT];
#pragma unroll
    for (int i = 0; i < MT; ++i) {
        const int row_c = imin((tm0 + i) * 16 + r, M - 1);
        aoff[i] = (size_t)row_c * K + (g << 3);
    }
    const size_t boff = (size_t)(g << 3) * N + col_c;

    v8f acc[MT];
#pragma unroll
    for (int i = 0; i < MT; ++i) acc[i] = (v8f){};

    for (int k0 = 0; k0 < K; k0 += 32) {
        // ---- B: lane holds col col_c, K slots {g*8+0..7, 16+g*8+0..7} (stride N) ----
        const size_t bk = boff + (size_t)k0 * N;
        v16h b;
#pragma unroll
        for (int t = 0; t < 16; ++t)
            b[t] = (_Float16)W[bk + (size_t)(t + ((t >> 3) << 3)) * N];
        // ---- A: lane holds its row, K slots {g*8+0..7, 16+g*8+0..7} ----
        v16h a[MT];
#pragma unroll
        for (int i = 0; i < MT; ++i) {
            const float* __restrict__ ai = X + aoff[i] + k0;
            if (k0 + 32 < K)
                __builtin_prefetch(ai + 32, 0, 0);
            const float4 c0 = *(const float4*)(ai + 0);
            const float4 c1 = *(const float4*)(ai + 4);
            const float4 c2 = *(const float4*)(ai + 16);
            const float4 c3 = *(const float4*)(ai + 20);
            a[i][0]  = (_Float16)c0.x; a[i][1]  = (_Float16)c0.y;
            a[i][2]  = (_Float16)c0.z; a[i][3]  = (_Float16)c0.w;
            a[i][4]  = (_Float16)c1.x; a[i][5]  = (_Float16)c1.y;
            a[i][6]  = (_Float16)c1.z; a[i][7]  = (_Float16)c1.w;
            a[i][8]  = (_Float16)c2.x; a[i][9]  = (_Float16)c2.y;
            a[i][10] = (_Float16)c2.z; a[i][11] = (_Float16)c2.w;
            a[i][12] = (_Float16)c3.x; a[i][13] = (_Float16)c3.y;
            a[i][14] = (_Float16)c3.z; a[i][15] = (_Float16)c3.w;
        }
#pragma unroll
        for (int i = 0; i < MT; ++i)
            acc[i] = __builtin_amdgcn_wmma_f32_16x16x32_f16(
                false, a[i], false, b, (short)0, acc[i], false, false);
    }
#pragma unroll
    for (int i = 0; i < MT; ++i) {
#pragma unroll
        for (int v = 0; v < 8; ++v) {
            const int m = (tm0 + i) * 16 + v + (g << 3); // D: VGPR v -> M=v+8g
            if (m < M && col < N) {
                float val = acc[i][v];
                if (bias)  val += bias[col];
                if (ACT)   val = val > 0.0f ? val : 0.0f;
                if (resid) val += resid[(size_t)m * N + col];
                Y[(size_t)m * N + col] = val;
            }
        }
    }
}

// ---------------------------------------------------------------------------
// Generic WMMA GEMM (any K, e.g. K=3): branch-free via clamped unconditional
// loads + v_cndmask zeroing of K-tail elements.
// ---------------------------------------------------------------------------
template <int ACT>
__global__ void wmma_gemm_gen_kernel(const float* __restrict__ X,
                                     const float* __restrict__ W,
                                     const float* __restrict__ bias,
                                     const float* __restrict__ resid,
                                     float* __restrict__ Y,
                                     int M, int K, int N)
{
    const int lane = threadIdx.x & 31;
    const int wave = threadIdx.x >> 5;
    const int tn   = blockIdx.x;
    const int tm   = blockIdx.y * 4 + wave;
    const int g    = lane >> 4;
    const int r    = lane & 15;
    const int row  = tm * 16 + r;
    const int col  = tn * 16 + r;
    const int row_c = imin(row, M - 1);
    const int col_c = imin(col, N - 1);
    const size_t xoff = (size_t)row_c * K;

    v8f acc = {};
    for (int k0 = 0; k0 < K; k0 += 32) {
        v16h a, b;
#pragma unroll
        for (int v = 0; v < 8; ++v) {
#pragma unroll
            for (int s = 0; s < 2; ++s) {
                const int ka = k0 + ((v & 3) << 1) + s + (g << 3) + ((v >> 2) << 4);
                const int kb = k0 + ((v & 3) << 1) + s + (g << 3) + ((v >> 2) << 4);
                float av = X[xoff + imin(ka, K - 1)];        // unconditional load
                float bv = W[(size_t)imin(kb, K - 1) * N + col_c];
                av = (ka < K) ? av : 0.0f;                   // cndmask, no branch
                bv = (kb < K) ? bv : 0.0f;
                a[v * 2 + s] = (_Float16)av;
                b[v * 2 + s] = (_Float16)bv;
            }
        }
        acc = __builtin_amdgcn_wmma_f32_16x16x32_f16(
            false, a, false, b, (short)0, acc, false, false);
    }
#pragma unroll
    for (int v = 0; v < 8; ++v) {
        const int m = tm * 16 + v + (g << 3);
        if (m < M && col < N) {
            float val = acc[v];
            if (bias)  val += bias[col];
            if (ACT)   val = val > 0.0f ? val : 0.0f;
            if (resid) val += resid[(size_t)m * N + col];
            Y[(size_t)m * N + col] = val;
        }
    }
}

// ---------------------------------------------------------------------------
// KNN: one thread per query point, insertion top-k (k <= 16), stable ties.
// ---------------------------------------------------------------------------
__global__ void knn_kernel(const float* __restrict__ q, const float* __restrict__ ref,
                           int* __restrict__ out, int nq, int nr, int k)
{
    int t = blockIdx.x * blockDim.x + threadIdx.x;
    if (t >= BB * nq) return;
    const int b = t / nq;
    const float* qp = q + (size_t)t * 3;
    const float* rp = ref + (size_t)b * nr * 3;
    const float qx = qp[0], qy = qp[1], qz = qp[2];
    float bd[16]; int bi[16];
    for (int j = 0; j < k; ++j) { bd[j] = 3.0e38f; bi[j] = 0; }
    for (int i = 0; i < nr; ++i) {
        float dx = rp[i * 3 + 0] - qx;
        float dy = rp[i * 3 + 1] - qy;
        float dz = rp[i * 3 + 2] - qz;
        float d = dx * dx + dy * dy + dz * dz;
        if (d < bd[k - 1]) {
            int p = k - 1;
            while (p > 0 && bd[p - 1] > d) { bd[p] = bd[p - 1]; bi[p] = bi[p - 1]; --p; }
            bd[p] = d; bi[p] = i;
        }
    }
    for (int j = 0; j < k; ++j) out[(size_t)t * k + j] = bi[j];
}

// ---------------------------------------------------------------------------
// Farthest point sampling: one block per batch; LDS dist[] + argmax reduce.
// Matches the reference scan: cents[0] = 0; cents[i] = argmax after i updates.
// ---------------------------------------------------------------------------
__global__ void fps_kernel(const float* __restrict__ xyz, int* __restrict__ out,
                           int n, int npoint)
{
    __shared__ float dist[NN];
    __shared__ float rv[256];
    __shared__ int   ri[256];
    const int b = blockIdx.x;
    const int t = threadIdx.x;
    const float* p = xyz + (size_t)b * n * 3;
    for (int i = t; i < n; i += 256) dist[i] = 1.0e10f;
    __syncthreads();
    int far = 0;
    for (int it = 0; it < npoint; ++it) {
        if (t == 0) out[b * npoint + it] = far;
        const float cx = p[far * 3 + 0], cy = p[far * 3 + 1], cz = p[far * 3 + 2];
        float bv = -1.0f; int bi = 0;
        for (int i = t; i < n; i += 256) {
            float dx = p[i * 3 + 0] - cx, dy = p[i * 3 + 1] - cy, dz = p[i * 3 + 2] - cz;
            float d = fminf(dist[i], dx * dx + dy * dy + dz * dz);
            dist[i] = d;
            if (d > bv) { bv = d; bi = i; }   // strided i increases -> first-index ties
        }
        rv[t] = bv; ri[t] = bi;
        __syncthreads();
        for (int s = 128; s > 0; s >>= 1) {
            if (t < s) {
                if (rv[t + s] > rv[t] || (rv[t + s] == rv[t] && ri[t + s] < ri[t])) {
                    rv[t] = rv[t + s]; ri[t] = ri[t + s];
                }
            }
            __syncthreads();
        }
        far = ri[0];
        __syncthreads();
    }
}

// ---------------------------------------------------------------------------
// Gather rows: dst[b, j, :] = src[b, idx[b, j], :]
// ---------------------------------------------------------------------------
__global__ void gather_rows_kernel(const float* __restrict__ src, const int* __restrict__ idx,
                                   float* __restrict__ dst, int n_src, int n_idx, int C)
{
    size_t t = (size_t)blockIdx.x * blockDim.x + threadIdx.x;
    size_t total = (size_t)BB * n_idx * C;
    if (t >= total) return;
    int c = (int)(t % C);
    size_t r = t / C;
    int b = (int)(r / n_idx);
    dst[t] = src[((size_t)b * n_src + idx[r]) * C + c];
}

// pos input: out[b,m,j,:] = xyz[b,m,:] - xyz[b, idx[b,m,j], :]
__global__ void posdiff_kernel(const float* __restrict__ xyz, const int* __restrict__ idx,
                               float* __restrict__ out, int n, int k)
{
    int t = blockIdx.x * blockDim.x + threadIdx.x;
    int total = BB * n * k;
    if (t >= total) return;
    int bm = t / k;
    int b = bm / n;
    int nb = idx[t];
    const float* pq = xyz + (size_t)bm * 3;
    const float* pr = xyz + ((size_t)b * n + nb) * 3;
    out[(size_t)t * 3 + 0] = pq[0] - pr[0];
    out[(size_t)t * 3 + 1] = pq[1] - pr[1];
    out[(size_t)t * 3 + 2] = pq[2] - pr[2];
}

// attn_in[b,m,j,f] = q[b,m,f] - psix[b, idx[b,m,j], f] + pos[b,m,j,f]
__global__ void attn_in_kernel(const float* __restrict__ q, const float* __restrict__ psix,
                               const float* __restrict__ pos, const int* __restrict__ idx,
                               float* __restrict__ out, int n, int k, int F)
{
    size_t t = (size_t)blockIdx.x * blockDim.x + threadIdx.x;
    size_t total = (size_t)BB * n * k * F;
    if (t >= total) return;
    int f = (int)(t % F);
    size_t rj = t / F;            // bm*k + j
    size_t bm = rj / k;
    int b = (int)(bm / n);
    out[t] = q[bm * F + f] - psix[((size_t)b * n + idx[rj]) * F + f] + pos[t];
}

// Fused: softmax over k (axis=-2, scale 1/sqrt(F)) then weighted sum of
// (alphax[gather] + pos). One thread per (b,m,f).
__global__ void softmax_wsum_kernel(const float* __restrict__ attn,
                                    const float* __restrict__ alphax,
                                    const float* __restrict__ pos,
                                    const int* __restrict__ idx,
                                    float* __restrict__ res, int n, int k, int F)
{
    size_t t = (size_t)blockIdx.x * blockDim.x + threadIdx.x;
    size_t total = (size_t)BB * n * F;
    if (t >= total) return;
    int f = (int)(t % F);
    size_t bm = t / F;
    int b = (int)(bm / n);
    const float sc = rsqrtf((float)F);
    float mx = -3.0e38f;
    for (int j = 0; j < k; ++j)
        mx = fmaxf(mx, attn[(bm * k + j) * F + f] * sc);
    float se = 0.0f;
    for (int j = 0; j < k; ++j)
        se += expf(attn[(bm * k + j) * F + f] * sc - mx);
    const float inv = 1.0f / se;
    float acc = 0.0f;
    for (int j = 0; j < k; ++j) {
        size_t rj = bm * k + j;
        float w = expf(attn[rj * F + f] * sc - mx) * inv;
        acc += w * (alphax[((size_t)b * n + idx[rj]) * F + f] + pos[rj * F + f]);
    }
    res[t] = acc;
}

// ---------------------------------------------------------------------------
// BatchNorm over all-but-channel: LDS-atomic partial sums, then fused
// normalize + scale/shift + ReLU (in-place).
// ---------------------------------------------------------------------------
__global__ void bn_reduce_kernel(const float* __restrict__ x, float* __restrict__ sums,
                                 int M, int C)
{
    __shared__ float ls[2 * DM];
    for (int i = threadIdx.x; i < 2 * C; i += blockDim.x) ls[i] = 0.0f;
    __syncthreads();
    size_t total = (size_t)M * C;
    for (size_t i = (size_t)blockIdx.x * blockDim.x + threadIdx.x; i < total;
         i += (size_t)gridDim.x * blockDim.x) {
        int c = (int)(i % C);
        float v = x[i];
        atomicAdd(&ls[c], v);
        atomicAdd(&ls[C + c], v * v);
    }
    __syncthreads();
    for (int i = threadIdx.x; i < 2 * C; i += blockDim.x) atomicAdd(&sums[i], ls[i]);
}

__global__ void bn_apply_relu_kernel(float* __restrict__ x, const float* __restrict__ sums,
                                     const float* __restrict__ gam, const float* __restrict__ bet,
                                     int M, int C)
{
    size_t t = (size_t)blockIdx.x * blockDim.x + threadIdx.x;
    if (t >= (size_t)M * C) return;
    int c = (int)(t % C);
    float cnt = (float)M;
    float mu = sums[c] / cnt;
    float var = sums[C + c] / cnt - mu * mu;
    float y = gam[c] * (x[t] - mu) * rsqrtf(var + 1e-5f) + bet[c];
    x[t] = y > 0.0f ? y : 0.0f;
}

__global__ void maxpool_kernel(const float* __restrict__ h, float* __restrict__ out,
                               int Mo, int k, int C)
{
    size_t t = (size_t)blockIdx.x * blockDim.x + threadIdx.x;
    if (t >= (size_t)Mo * C) return;
    int c = (int)(t % C);
    size_t mo = t / C;
    float m = -3.0e38f;
    for (int j = 0; j < k; ++j) m = fmaxf(m, h[((size_t)mo * k + j) * C + c]);
    out[t] = m;
}

__global__ void mean_kernel(const float* __restrict__ f, float* __restrict__ out, int n, int C)
{
    int t = blockIdx.x * blockDim.x + threadIdx.x;
    if (t >= BB * C) return;
    int c = t % C;
    int b = t / C;
    float s = 0.0f;
    for (int m = 0; m < n; ++m) s += f[((size_t)b * n + m) * C + c];
    out[t] = s / (float)n;
}

// ---------------------------------------------------------------------------
// Host side
// ---------------------------------------------------------------------------
static inline void gemm(hipStream_t s, const float* X, const float* W, const float* b,
                        const float* r, float* Y, int M, int K, int N, bool relu)
{
    const bool fast = (K >= 32) && ((K & 31) == 0);
    if (fast) {
        if (M >= 256) {       // M-register-blocked: 4 tiles/wave, 256 rows/block
            dim3 grid(CDIVU(N, 16), CDIVU(M, 256));
            if (relu) wmma_gemm_fast_kernel<1, 4><<<grid, 128, 0, s>>>(X, W, b, r, Y, M, K, N);
            else      wmma_gemm_fast_kernel<0, 4><<<grid, 128, 0, s>>>(X, W, b, r, Y, M, K, N);
        } else {
            dim3 grid(CDIVU(N, 16), CDIVU(M, 64));
            if (relu) wmma_gemm_fast_kernel<1, 1><<<grid, 128, 0, s>>>(X, W, b, r, Y, M, K, N);
            else      wmma_gemm_fast_kernel<0, 1><<<grid, 128, 0, s>>>(X, W, b, r, Y, M, K, N);
        }
    } else {
        dim3 grid(CDIVU(N, 16), CDIVU(M, 64));
        if (relu) wmma_gemm_gen_kernel<1><<<grid, 128, 0, s>>>(X, W, b, r, Y, M, K, N);
        else      wmma_gemm_gen_kernel<0><<<grid, 128, 0, s>>>(X, W, b, r, Y, M, K, N);
    }
}

struct Blk {
    const float *f1w, *f1b, *f2w, *f2b;
    const float *d1w, *d1b, *d2w, *d2b;
    const float *g1w, *g1b, *g2w, *g2b;
    const float *phw, *psw, *alw;
};
struct Td {
    const float *c1w, *c1b, *g1, *b1, *c2w, *c2b, *g2, *b2;
};

extern "C" void kernel_launch(void* const* d_in, const int* in_sizes, int n_in,
                              void* d_out, int out_size, void* d_ws, size_t ws_size,
                              hipStream_t stream)
{
    (void)in_sizes; (void)n_in; (void)out_size; (void)ws_size;

    // ---- parse inputs (setup_inputs insertion order) ----
    const float* x = (const float*)d_in[0];
    int ip = 1;
    auto nxt = [&]() { return (const float*)d_in[ip++]; };
    auto readBlk = [&]() {
        Blk b;
        b.f1w = nxt(); b.f1b = nxt();
        b.f2w = nxt(); b.f2b = nxt();
        b.d1w = nxt(); b.d1b = nxt(); b.d2w = nxt(); b.d2b = nxt();
        b.g1w = nxt(); b.g1b = nxt(); b.g2w = nxt(); b.g2b = nxt();
        b.phw = nxt(); b.psw = nxt(); b.alw = nxt();
        return b;
    };
    const float *fc1w0 = nxt(), *fc1b0 = nxt(), *fc1w1 = nxt(), *fc1b1 = nxt();
    Blk t1p = readBlk();
    Td tds[4];
    for (int i = 0; i < 4; ++i) {
        tds[i].c1w = nxt(); tds[i].c1b = nxt();
        tds[i].g1  = nxt(); tds[i].b1  = nxt();
        tds[i].c2w = nxt(); tds[i].c2b = nxt();
        tds[i].g2  = nxt(); tds[i].b2  = nxt();
    }
    Blk tfs[4];
    for (int i = 0; i < 4; ++i) tfs[i] = readBlk();
    const float *fcw0 = nxt(), *fcb0 = nxt(), *fcw1 = nxt(), *fcb1 = nxt(),
                *fcw2 = nxt(), *fcb2 = nxt();

    // ---- workspace carve (bump allocator, 256B aligned) ----
    char* wp = (char*)d_ws;
    auto allocf = [&](size_t nelem) {
        float* r = (float*)wp;
        wp += ((nelem * sizeof(float)) + 255) & ~(size_t)255;
        return r;
    };
    auto alloci = [&](size_t nelem) {
        int* r = (int*)wp;
        wp += ((nelem * sizeof(int)) + 255) & ~(size_t)255;
        return r;
    };
    const size_t MR = (size_t)BB * NN;           // 4096 max rows
    float* xb     = allocf(MR * DM);
    float* qb     = allocf(MR * DM);
    float* psixb  = allocf(MR * DM);
    float* alphab = allocf(MR * DM);
    float* resb   = allocf(MR * DM);
    float* posd   = allocf(MR * 16 * 3);
    float* posB   = allocf(MR * 16 * DM);        // 128 MB
    float* T1     = allocf(MR * 16 * DM);        // 128 MB
    float* T2     = allocf(MR * 16 * DM);        // 128 MB
    float* bnsum  = allocf(2 * DM);
    int*   idxb   = alloci(MR * 16);
    int*   fidxb  = alloci(BB * 256);
    float* fbufs[5];
    fbufs[0] = allocf((size_t)BB * 1024 * 32);
    fbufs[1] = allocf((size_t)BB * 256 * 64);
    fbufs[2] = allocf((size_t)BB * 64 * 128);
    fbufs[3] = allocf((size_t)BB * 16 * 256);
    fbufs[4] = allocf((size_t)BB * 4 * 512);
    float* xyzbufs[4];
    xyzbufs[0] = allocf((size_t)BB * 256 * 3);
    xyzbufs[1] = allocf((size_t)BB * 64 * 3);
    xyzbufs[2] = allocf((size_t)BB * 16 * 3);
    xyzbufs[3] = allocf((size_t)BB * 4 * 3);
    float* pooled = allocf(BB * 512);
    float* h1     = allocf(BB * 256);
    float* h2     = allocf(BB * 64);

    // ---- point transformer block ----
    auto pt_block = [&](const Blk& P, const float* xyz, float* feats, int n, int k, int din) {
        const int Mr = BB * n;
        const int Mk = Mr * k;
        knn_kernel<<<CDIVU(Mr, 128), 128, 0, stream>>>(xyz, xyz, idxb, n, n, k);
        gemm(stream, feats, P.f1w, P.f1b, nullptr, xb,     Mr, din, DM, false);
        gemm(stream, xb,    P.phw, nullptr, nullptr, qb,     Mr, DM, DM, false);
        gemm(stream, xb,    P.psw, nullptr, nullptr, psixb,  Mr, DM, DM, false);
        gemm(stream, xb,    P.alw, nullptr, nullptr, alphab, Mr, DM, DM, false);
        posdiff_kernel<<<CDIVU(Mk, 256), 256, 0, stream>>>(xyz, idxb, posd, n, k);
        gemm(stream, posd, P.d1w, P.d1b, nullptr, T1,   Mk, 3,  DM, true);
        gemm(stream, T1,   P.d2w, P.d2b, nullptr, posB, Mk, DM, DM, false);
        attn_in_kernel<<<CDIVU((size_t)Mk * DM, 256), 256, 0, stream>>>(
            qb, psixb, posB, idxb, T1, n, k, DM);
        gemm(stream, T1, P.g1w, P.g1b, nullptr, T2, Mk, DM, DM, true);
        gemm(stream, T2, P.g2w, P.g2b, nullptr, T1, Mk, DM, DM, false);
        softmax_wsum_kernel<<<CDIVU((size_t)Mr * DM, 256), 256, 0, stream>>>(
            T1, alphab, posB, idxb, resb, n, k, DM);
        // out = lin(res, fc2) + feats  (in-place: each element read+written once)
        gemm(stream, resb, P.f2w, P.f2b, feats, feats, Mr, DM, din, false);
    };

    // ---- transition-down block ----
    auto td_step = [&](const Td& P, const float* xyz_in, const float* f_in,
                       float* xyz_out, float* f_out, int n, int npnt, int k, int cin) {
        const int cout = 2 * cin;
        fps_kernel<<<BB, 256, 0, stream>>>(xyz_in, fidxb, n, npnt);
        gather_rows_kernel<<<CDIVU((size_t)BB * npnt * 3, 256), 256, 0, stream>>>(
            xyz_in, fidxb, xyz_out, n, npnt, 3);
        knn_kernel<<<CDIVU(BB * npnt, 128), 128, 0, stream>>>(xyz_out, xyz_in, idxb, npnt, n, k);
        gather_rows_kernel<<<CDIVU((size_t)BB * npnt * k * cin, 256), 256, 0, stream>>>(
            f_in, idxb, posB, n, npnt * k, cin);
        const int Mh = BB * npnt * k;
        gemm(stream, posB, P.c1w, P.c1b, nullptr, T1, Mh, cin, cout, false);
        hipMemsetAsync(bnsum, 0, 2 * DM * sizeof(float), stream);
        bn_reduce_kernel<<<256, 256, 0, stream>>>(T1, bnsum, Mh, cout);
        bn_apply_relu_kernel<<<CDIVU((size_t)Mh * cout, 256), 256, 0, stream>>>(
            T1, bnsum, P.g1, P.b1, Mh, cout);
        gemm(stream, T1, P.c2w, P.c2b, nullptr, T2, Mh, cout, cout, false);
        hipMemsetAsync(bnsum, 0, 2 * DM * sizeof(float), stream);
        bn_reduce_kernel<<<256, 256, 0, stream>>>(T2, bnsum, Mh, cout);
        bn_apply_relu_kernel<<<CDIVU((size_t)Mh * cout, 256), 256, 0, stream>>>(
            T2, bnsum, P.g2, P.b2, Mh, cout);
        maxpool_kernel<<<CDIVU((size_t)BB * npnt * cout, 256), 256, 0, stream>>>(
            T2, f_out, BB * npnt, k, cout);
    };

    // ---- forward ----
    // f = mlp2(x, fc1):  (BN,3) -> relu(lin 3->32) -> lin 32->32
    gemm(stream, x,  fc1w0, fc1b0, nullptr, xb,       BB * NN, 3,  32, true);
    gemm(stream, xb, fc1w1, fc1b1, nullptr, fbufs[0], BB * NN, 32, 32, false);

    // first transformer block (xyz == x since D_IN == 3)
    pt_block(t1p, x, fbufs[0], NN, 16, 32);

    const float* xyzc = x;
    float* fcur = fbufs[0];
    int n = NN;
    for (int i = 0; i < 4; ++i) {
        const int npnt = NN >> (2 * (i + 1));        // 256, 64, 16, 4
        const int k_td = n < 16 ? n : 16;            // k uses pre-downsample n
        td_step(tds[i], xyzc, fcur, xyzbufs[i], fbufs[i + 1], n, npnt, k_td, 32 << i);
        xyzc = xyzbufs[i];
        fcur = fbufs[i + 1];
        n = npnt;
        const int k_tf = n < 16 ? n : 16;            // 16,16,16,4
        pt_block(tfs[i], xyzc, fcur, n, k_tf, 32 << (i + 1));
    }

    // global mean pool over points (n == 4, C == 512) and classifier head
    mean_kernel<<<CDIVU(BB * 512, 256), 256, 0, stream>>>(fcur, pooled, n, 512);
    gemm(stream, pooled, fcw0, fcb0, nullptr, h1, BB, 512, 256, true);
    gemm(stream, h1,     fcw1, fcb1, nullptr, h2, BB, 256, 64,  true);
    gemm(stream, h2,     fcw2, fcb2, nullptr, (float*)d_out, BB, 64, 40, false);
}
